// GINNet_37769942401055
// MI455X (gfx1250) — matrix operand order repaired and checked by reading, not compiled, not run on hardware
//
#include <hip/hip_runtime.h>
#include <hip/hip_bf16.h>

#define N_NODES    100000
#define N_PAD      100096            // padded to GEMM block extent (782 * 128)
#define N_EDGES    800000
#define HID        128
#define NUM_CLASSES 10
#define NUM_GRAPHS 128
#define BN_EPS     1e-5f

typedef __attribute__((ext_vector_type(16))) _Float16 v16h;
typedef __attribute__((ext_vector_type(8)))  _Float16 v8h;
typedef __attribute__((ext_vector_type(8)))  float    v8f;

union H16 { v16h v; v8h h[2]; };

// ---------------- elementwise helpers ----------------

__global__ void k_zero(float* __restrict__ p, long long n) {
  long long i = (long long)blockIdx.x * blockDim.x + threadIdx.x;
  if (i < n) p[i] = 0.0f;
}

// agg[dst] += feats[src], one thread per (edge, feature)
__global__ void k_scatter(const float* __restrict__ feats,
                          const int* __restrict__ esrc,
                          const int* __restrict__ edst,
                          float* __restrict__ agg) {
  long long t = (long long)blockIdx.x * blockDim.x + threadIdx.x;
  int e = (int)(t >> 7);
  int f = (int)(t & 127);
  if (e < N_EDGES) {
    int s = esrc[e], d = edst[e];
    atomicAdd(agg + (size_t)d * HID + f, feats[(size_t)s * HID + f]);
  }
}

// a16 = f16(x + agg)   (GIN eps = 0)
__global__ void k_build_a(const float* __restrict__ x,
                          const float* __restrict__ agg,
                          _Float16* __restrict__ a16, long long n) {
  long long i = (long long)blockIdx.x * blockDim.x + threadIdx.x;
  if (i < n) a16[i] = (_Float16)(x[i] + agg[i]);
}

// ---------------- WMMA GEMM: C[N_PAD x 128] = A * W + b ----------------
// All node buffers are padded to N_PAD rows -> no bounds guards anywhere:
// loads, WMMA, and stores run with EXEC all-ones, branch-free.
// EPI 0: relu + f16 out (feeds next GEMM).  EPI 1: f32 out (feeds BN).
template <int EPI>
__global__ __launch_bounds__(256)
void k_gemm(const _Float16* __restrict__ A,
            const float* __restrict__ W,      // [128][128] f32, row = k, col = n
            const float* __restrict__ bias,   // [128]
            _Float16* __restrict__ out16,
            float* __restrict__ out32) {
  // W transposed into LDS as f16 with pitch 136 halves (272 B) -> conflict-free b128 loads
  __shared__ _Float16 Wt[128 * 136];
  int tid = threadIdx.x;
  for (int i = tid; i < 128 * 128; i += 256) {
    int k = i >> 7, n = i & 127;
    Wt[n * 136 + k] = (_Float16)W[i];
  }
  __syncthreads();

  int wave = tid >> 5, lane = tid & 31;
  int rowbase = blockIdx.x * 128 + wave * 16;
  int mloc = lane & 15;     // row within 16-row strip (A), col within 16-col tile (B/C)
  int hsel = lane >> 4;     // half-wave select

  v8f acc[8] = {};

  const _Float16* abase = A + (size_t)(rowbase + mloc) * HID + hsel * 8;

#pragma unroll
  for (int kc = 0; kc < 4; ++kc) {
    // A fragment: 16-bit A 16x32 layout; lanes 16-31 hold K+8 / K+24 groups
    H16 af;
    af.h[0] = *(const v8h*)(abase + kc * 32);
    af.h[1] = *(const v8h*)(abase + kc * 32 + 16);
#pragma unroll
    for (int nt = 0; nt < 8; ++nt) {
      // B fragment: column n = nt*16+mloc, K groups 8*hsel and 16+8*hsel (matches A)
      const _Float16* bp = &Wt[(nt * 16 + mloc) * 136 + kc * 32 + hsel * 8];
      H16 bf;
      bf.h[0] = *(const v8h*)(bp);
      bf.h[1] = *(const v8h*)(bp + 16);
      acc[nt] = __builtin_amdgcn_wmma_f32_16x16x32_f16(
          false, af.v, false, bf.v, (short)0, acc[nt], false, false);
    }
  }

  // epilogue: C layout -> VGPR r holds row (r + 8*hsel), col = mloc of each tile
#pragma unroll
  for (int nt = 0; nt < 8; ++nt) {
    int n = nt * 16 + mloc;
    float b = bias[n];
#pragma unroll
    for (int r = 0; r < 8; ++r) {
      int m = rowbase + hsel * 8 + r;
      float v = acc[nt][r] + b;
      if (EPI == 0) {
        v = fmaxf(v, 0.0f);
        out16[(size_t)m * HID + n] = (_Float16)v;
      } else {
        out32[(size_t)m * HID + n] = v;
      }
    }
  }
}

// ---------------- BatchNorm (training-mode, biased var) ----------------

__global__ void k_bn_stats(const float* __restrict__ h, float* __restrict__ stats) {
  int f = threadIdx.x;               // 128 threads = 128 features
  int r0 = blockIdx.x * 512;
  int r1 = r0 + 512; if (r1 > N_NODES) r1 = N_NODES;
  float s = 0.0f, ss = 0.0f;
  for (int r = r0; r < r1; ++r) {
    float v = h[(size_t)r * HID + f];
    s += v; ss += v * v;
  }
  atomicAdd(stats + f, s);
  atomicAdd(stats + HID + f, ss);
}

__global__ void k_bn_apply(float* __restrict__ h, const float* __restrict__ stats,
                           const float* __restrict__ gamma, const float* __restrict__ beta) {
  long long i = (long long)blockIdx.x * blockDim.x + threadIdx.x;
  if (i >= (long long)N_NODES * HID) return;
  int f = (int)(i & 127);
  const float invN = 1.0f / (float)N_NODES;
  float mean = stats[f] * invN;
  float var  = stats[HID + f] * invN - mean * mean;
  float v = (h[i] - mean) * rsqrtf(var + BN_EPS) * gamma[f] + beta[f];
  h[i] = fmaxf(v, 0.0f);             // fused ReLU after BN
}

// ---------------- pooling + classifier head ----------------

__global__ void k_pool(const float* __restrict__ h, const int* __restrict__ batch,
                       float* __restrict__ pooled, float* __restrict__ counts) {
  long long i = (long long)blockIdx.x * blockDim.x + threadIdx.x;
  if (i >= (long long)N_NODES * HID) return;
  int r = (int)(i >> 7), f = (int)(i & 127);
  int g = batch[r];
  atomicAdd(pooled + (size_t)g * HID + f, h[i]);
  if (f == 0) atomicAdd(counts + g, 1.0f);
}

__global__ void k_head(const float* __restrict__ pooled, const float* __restrict__ counts,
                       const float* __restrict__ Wlin, const float* __restrict__ blin,
                       float* __restrict__ out) {
  __shared__ float logit[NUM_CLASSES];
  int g = blockIdx.x;
  int t = threadIdx.x;               // 32 threads (one wave32)
  float inv = 1.0f / fmaxf(counts[g], 1.0f);
  if (t < NUM_CLASSES) {
    float acc = blin[t];
    for (int k = 0; k < HID; ++k)
      acc += pooled[(size_t)g * HID + k] * inv * Wlin[k * NUM_CLASSES + t];
    logit[t] = acc;
  }
  __syncthreads();
  if (t < NUM_CLASSES) {
    float mx = -1e30f;
    for (int c = 0; c < NUM_CLASSES; ++c) mx = fmaxf(mx, logit[c]);
    float s = 0.0f;
    for (int c = 0; c < NUM_CLASSES; ++c) s += expf(logit[c] - mx);
    out[g * NUM_CLASSES + t] = logit[t] - mx - logf(s);
  }
}

// ---------------- launcher ----------------

extern "C" void kernel_launch(void* const* d_in, const int* in_sizes, int n_in,
                              void* d_out, int out_size, void* d_ws, size_t ws_size,
                              hipStream_t stream) {
  const float* x     = (const float*)d_in[0];
  const int*   eidx  = (const int*)d_in[1];
  const int*   esrc  = eidx;
  const int*   edst  = eidx + N_EDGES;
  const int*   batch = (const int*)d_in[2];
  const float* W1a = (const float*)d_in[3],  *b1a = (const float*)d_in[4];
  const float* W1b = (const float*)d_in[5],  *b1b = (const float*)d_in[6];
  const float* g1  = (const float*)d_in[7],  *bt1 = (const float*)d_in[8];
  const float* W2a = (const float*)d_in[9],  *b2a = (const float*)d_in[10];
  const float* W2b = (const float*)d_in[11], *b2b = (const float*)d_in[12];
  const float* g2  = (const float*)d_in[13], *bt2 = (const float*)d_in[14];
  const float* Wln = (const float*)d_in[15], *bln = (const float*)d_in[16];
  float* out = (float*)d_out;

  const long long NE    = (long long)N_NODES * HID;    // 12.8M valid elems
  const long long NEP   = (long long)N_PAD * HID;      // padded elems
  const size_t F32_B    = (size_t)NEP * sizeof(float); // padded f32 buffer bytes
  const size_t F16_B    = (size_t)NEP * 2;             // padded f16 buffer bytes
  const long long PAD_F = (NEP - NE) / 2;              // f16 pad region as float count

  char* ws = (char*)d_ws;
  float*    agg    = (float*)(ws);
  float*    hf     = (float*)(ws + F32_B);
  _Float16* a16    = (_Float16*)(ws + 2 * F32_B);
  _Float16* m16    = (_Float16*)(ws + 2 * F32_B + F16_B);
  float*    stats  = (float*)(ws + 2 * F32_B + 2 * F16_B);       // 256 f32
  float*    pooled = stats + 2 * HID;                            // 128*128 f32
  float*    counts = pooled + NUM_GRAPHS * HID;                  // 128 f32

  dim3 blk(256);
  unsigned gEW    = (unsigned)((NE + 255) / 256);                      // elementwise over valid nodes*feats
  unsigned gSCAT  = (unsigned)(((long long)N_EDGES * HID + 255) / 256);
  unsigned gGEMM  = N_PAD / 128;                                       // 782, exact
  unsigned gSTATS = (N_NODES + 511) / 512;

  // zero stats + pooled + counts region once
  k_zero<<<(2 * HID + NUM_GRAPHS * HID + NUM_GRAPHS + 255) / 256, blk, 0, stream>>>(
      stats, 2 * HID + NUM_GRAPHS * HID + NUM_GRAPHS);
  // zero the f16 pad rows of a16/m16 once (keeps pad math finite & runs deterministic)
  k_zero<<<(unsigned)((PAD_F + 255) / 256), blk, 0, stream>>>((float*)(a16 + NE), PAD_F);
  k_zero<<<(unsigned)((PAD_F + 255) / 256), blk, 0, stream>>>((float*)(m16 + NE), PAD_F);

  // -------- conv1 --------
  k_zero   <<<gEW,   blk, 0, stream>>>(agg, NE);
  k_scatter<<<gSCAT, blk, 0, stream>>>(x, esrc, edst, agg);
  k_build_a<<<gEW,   blk, 0, stream>>>(x, agg, a16, NE);
  k_gemm<0><<<gGEMM, blk, 0, stream>>>(a16, W1a, b1a, m16, nullptr);
  k_gemm<1><<<gGEMM, blk, 0, stream>>>(m16, W1b, b1b, nullptr, hf);
  k_bn_stats<<<gSTATS, dim3(HID), 0, stream>>>(hf, stats);
  k_bn_apply<<<gEW, blk, 0, stream>>>(hf, stats, g1, bt1);

  // -------- conv2 --------
  k_zero   <<<gEW,   blk, 0, stream>>>(agg, NE);
  k_zero   <<<1,     blk, 0, stream>>>(stats, 2 * HID);
  k_scatter<<<gSCAT, blk, 0, stream>>>(hf, esrc, edst, agg);
  k_build_a<<<gEW,   blk, 0, stream>>>(hf, agg, a16, NE);
  k_gemm<0><<<gGEMM, blk, 0, stream>>>(a16, W2a, b2a, m16, nullptr);
  k_gemm<1><<<gGEMM, blk, 0, stream>>>(m16, W2b, b2b, nullptr, hf);
  k_bn_stats<<<gSTATS, dim3(HID), 0, stream>>>(hf, stats);
  k_bn_apply<<<gEW, blk, 0, stream>>>(hf, stats, g2, bt2);

  // -------- pool + head --------
  k_pool<<<gEW, blk, 0, stream>>>(hf, batch, pooled, counts);
  k_head<<<NUM_GRAPHS, dim3(32), 0, stream>>>(pooled, counts, Wln, bln, out);
}
